// BiLSTM_44074954391591
// MI455X (gfx1250) — matrix operand order repaired and checked by reading, not compiled, not run on hardware
//
#include <hip/hip_runtime.h>

// BiLSTM (TF BasicLSTMCell gate order i,j,f,o; forget_bias=1.0)
// B=64, T=2048, F=128, U=128.  out[B][T][2U] = concat(fw, bw).
//
// 8 workgroups (4 fw + 4 bw), each owns 16 batch rows on its own WGP; the
// recurrence is batch-parallel so there is zero cross-WG sync. W (256x512)
// is staged once into LDS as transposed bf16, then each wave hoists its 32
// loop-invariant B fragments (4 gates x 8 k-steps = 256 VGPRs) into registers
// for all 2048 timesteps -> the inner loop issues 32 register-fed
// v_wmma_f32_16x16x32_bf16 per wave with only 64 A-dwords per lane of LDS traffic.
// Cell state c lives in VGPRs; h is written back to LDS as the recurrent
// half of the A panel (two s_barriers per step).

typedef __attribute__((ext_vector_type(16))) __bf16 v16bf;
typedef __attribute__((ext_vector_type(8)))  float  v8f;

#define T_LEN 2048
#define F_DIM 128
#define U_DIM 128
#define NQ    512        // 4U gate columns
#define K_DIM 256        // F + U
#define PITCH 264        // LDS row pitch (bf16 elems); 528B rows stagger banks

union FragA { v16bf v; unsigned u[8]; };
union FragB { v16bf v; uint4 q[2]; };

__device__ __forceinline__ unsigned short f2bf(float f) {
  unsigned u = __builtin_bit_cast(unsigned, f);
  u += 0x7FFFu + ((u >> 16) & 1u);           // round-to-nearest-even
  return (unsigned short)(u >> 16);
}

__device__ __forceinline__ float sigm(float x) {
  return 1.0f / (1.0f + __expf(-x));
}
__device__ __forceinline__ float tanh_fast(float x) {
  return 1.0f - 2.0f / (__expf(2.0f * x) + 1.0f);
}

__global__ __launch_bounds__(256, 1) void bilstm_kernel(
    const float* __restrict__ x,
    const float* __restrict__ W_fw, const float* __restrict__ b_fw,
    const float* __restrict__ W_bw, const float* __restrict__ b_bw,
    float* __restrict__ out)
{
  extern __shared__ unsigned short smem[];
  unsigned short* Wt = smem;               // [NQ][PITCH]  W^T in bf16: Wt[n][k]
  unsigned short* xh = smem + NQ * PITCH;  // [16][PITCH]  A rows: [x_t | h_{t-1}]

  const int tid  = threadIdx.x;
  const int wave = tid >> 5;
  const int lane = tid & 31;
  const int nl   = lane & 15;              // n-within-tile == A row m (per layout)
  const int half = lane >> 4;

  const int dir = blockIdx.x >> 2;         // 0 = forward, 1 = backward
  const int b0  = (blockIdx.x & 3) << 4;   // 16 batch rows per WG

  const float* W    = dir ? W_bw : W_fw;
  const float* bias = dir ? b_bw : b_fw;

  // ---- one-time: W -> LDS, transposed bf16 (coalesced global reads) ----
  for (int idx = tid; idx < K_DIM * NQ; idx += 256) {
    const int k = idx >> 9;                // row of W (input dim)
    const int n = idx & (NQ - 1);          // gate column
    Wt[n * PITCH + k] = f2bf(W[idx]);
  }
  // zero recurrent half of A panel (h_{-1} = 0)
  for (int idx = tid; idx < 16 * U_DIM; idx += 256) {
    xh[(idx >> 7) * PITCH + F_DIM + (idx & 127)] = 0;
  }

  // per-lane hidden unit u and its four gate biases (forget bias folded)
  const int u = (wave << 4) + nl;
  const float bi  = bias[u];
  const float bj  = bias[U_DIM + u];
  const float bff = bias[2 * U_DIM + u] + 1.0f;
  const float bo  = bias[3 * U_DIM + u];

  const int t0 = dir ? (T_LEN - 1) : 0;
  const int dt = dir ? -1 : 1;

  // x staging: thread -> (row pm, 8 floats at col pf)
  const int pm = tid >> 4;
  const int pf = (tid & 15) << 3;
  const float* xbase = x + (size_t)(b0 + pm) * T_LEN * F_DIM + pf;

  {  // preload x(t0) into the A panel
    const float* p = xbase + (size_t)t0 * F_DIM;
    const float4 r0 = *(const float4*)p;
    const float4 r1 = *(const float4*)(p + 4);
    uint4 pk;
    pk.x = (unsigned)f2bf(r0.x) | ((unsigned)f2bf(r0.y) << 16);
    pk.y = (unsigned)f2bf(r0.z) | ((unsigned)f2bf(r0.w) << 16);
    pk.z = (unsigned)f2bf(r1.x) | ((unsigned)f2bf(r1.y) << 16);
    pk.w = (unsigned)f2bf(r1.z) | ((unsigned)f2bf(r1.w) << 16);
    *(uint4*)(xh + pm * PITCH + pf) = pk;
  }

  float cst[8];                            // cell state, resident in VGPRs
  #pragma unroll
  for (int r = 0; r < 8; ++r) cst[r] = 0.0f;

  // wave w handles N-tiles {g*8+w}: columns [16w,16w+16) of each gate g
  int brow[4];
  #pragma unroll
  for (int g = 0; g < 4; ++g)
    brow[g] = ((((g << 3) + wave) << 4) + nl) * PITCH + (half << 3);

  const unsigned* arow = (const unsigned*)(xh + nl * PITCH);
  const int ka0 = half << 2;               // half*8 bf16 => half*4 dwords

  __syncthreads();

  // ---- hoist all 32 loop-invariant B fragments into VGPRs (256 VGPRs) ----
  v16bf wb[32];
  #pragma unroll
  for (int ks = 0; ks < 8; ++ks) {
    #pragma unroll
    for (int g = 0; g < 4; ++g) {
      FragB t;
      const uint4* p = (const uint4*)(Wt + brow[g] + (ks << 5));
      t.q[0] = p[0];
      t.q[1] = p[2];
      wb[(ks << 2) + g] = t.v;
    }
  }

  for (int s = 0; s < T_LEN; ++s) {
    const int t = t0 + dt * s;

    // ---- z = [x_t | h] @ W : 8 K-steps, all B operands register-resident ----
    v8f acc[4] = {};
    #pragma unroll
    for (int ks = 0; ks < 8; ++ks) {
      FragA fa;                            // A frag: K = ks*32 + {0..7,16..23} + half*8
      const int ka = (ks << 4) + ka0;
      fa.u[0] = arow[ka + 0];
      fa.u[1] = arow[ka + 1];
      fa.u[2] = arow[ka + 2];
      fa.u[3] = arow[ka + 3];
      fa.u[4] = arow[ka + 8];
      fa.u[5] = arow[ka + 9];
      fa.u[6] = arow[ka + 10];
      fa.u[7] = arow[ka + 11];
      #pragma unroll
      for (int g = 0; g < 4; ++g) {
        acc[g] = __builtin_amdgcn_wmma_f32_16x16x32_bf16(
            false, fa.v, false, wb[(ks << 2) + g], (short)0, acc[g], false, false);
      }
    }

    // ---- prefetch x(t+dt) to regs; hint x(t+2dt) toward the caches ----
    float4 r0 = make_float4(0.f, 0.f, 0.f, 0.f);
    float4 r1 = make_float4(0.f, 0.f, 0.f, 0.f);
    if (s + 1 < T_LEN) {
      const float* p = xbase + (size_t)(t + dt) * F_DIM;
      r0 = *(const float4*)p;
      r1 = *(const float4*)(p + 4);
    }
    if (s + 2 < T_LEN) {
      __builtin_prefetch(xbase + (size_t)(t + 2 * dt) * F_DIM, 0, 0);  // global_prefetch_b8
    }

    __syncthreads();  // all reads of xh done before overwrite

    // ---- LSTM cell update; C layout: VGPR r -> row m = r + 8*half ----
    #pragma unroll
    for (int r = 0; r < 8; ++r) {
      const int m = r + (half << 3);
      const float zi = acc[0][r] + bi;
      const float zj = acc[1][r] + bj;
      const float zf = acc[2][r] + bff;
      const float zo = acc[3][r] + bo;
      const float cn = cst[r] * sigm(zf) + sigm(zi) * tanh_fast(zj);
      const float hn = sigm(zo) * tanh_fast(cn);
      cst[r] = cn;
      xh[m * PITCH + F_DIM + u] = f2bf(hn);                       // recurrent input
      out[((size_t)(b0 + m) * T_LEN + t) * (2 * U_DIM) + dir * U_DIM + u] = hn;
    }

    {  // commit prefetched x(t+dt) to the A panel
      uint4 pk;
      pk.x = (unsigned)f2bf(r0.x) | ((unsigned)f2bf(r0.y) << 16);
      pk.y = (unsigned)f2bf(r0.z) | ((unsigned)f2bf(r0.w) << 16);
      pk.z = (unsigned)f2bf(r1.x) | ((unsigned)f2bf(r1.y) << 16);
      pk.w = (unsigned)f2bf(r1.z) | ((unsigned)f2bf(r1.w) << 16);
      *(uint4*)(xh + pm * PITCH + pf) = pk;
    }

    __syncthreads();  // xh (x and h halves) visible for next step
  }
}

extern "C" void kernel_launch(void* const* d_in, const int* in_sizes, int n_in,
                              void* d_out, int out_size, void* d_ws, size_t ws_size,
                              hipStream_t stream) {
  const float* x    = (const float*)d_in[0];
  const float* W_fw = (const float*)d_in[1];
  const float* b_fw = (const float*)d_in[2];
  const float* W_bw = (const float*)d_in[3];
  const float* b_bw = (const float*)d_in[4];
  float* out = (float*)d_out;
  (void)in_sizes; (void)n_in; (void)out_size; (void)d_ws; (void)ws_size;

  const size_t smem = (size_t)(NQ + 16) * PITCH * sizeof(unsigned short); // ~272 KB
  hipFuncSetAttribute(reinterpret_cast<const void*>(bilstm_kernel),
                      hipFuncAttributeMaxDynamicSharedMemorySize, (int)smem);
  bilstm_kernel<<<dim3(8), dim3(256), smem, stream>>>(x, W_fw, b_fw, W_bw, b_bw, out);
}
